// BackwardPass_29137058136068
// MI455X (gfx1250) — matrix operand order repaired and checked by reading, not compiled, not run on hardware
//
#include <hip/hip_runtime.h>
#include <hip/hip_bf16.h>

typedef __attribute__((ext_vector_type(16))) _Float16 v16h;
typedef __attribute__((ext_vector_type(8)))  float    v8f;

#define D_MODEL 1024
#define HD      64      // head dim (DK)
#define BLK     128
#define TBLK    16
#define SCALE   0.125f  // TAU / sqrt(64)

// ---------------------------------------------------------------------------
// WMMA wrapper: D(16x16,f32) = A(16x32,f16) * B(32x16,f16) + C
// ---------------------------------------------------------------------------
__device__ __forceinline__ v8f wmma16(v16h a, v16h b, v8f c) {
  return __builtin_amdgcn_wmma_f32_16x16x32_f16(false, a, false, b, (short)0, c,
                                                false, false);
}

// A-operand (16x32 f16) from row-major f16 LDS tile, row stride ld (halves).
// Per ISA layout: lane m (m<16) holds row m, K {k0..k0+7, k0+16..k0+23};
// lane m+16 holds row m, K {k0+8..k0+15, k0+24..k0+31}.
__device__ __forceinline__ v16h load_a_tile(const _Float16* s, int row0, int k0,
                                            int ld, int lane) {
  int r   = row0 + (lane & 15);
  int off = (lane < 16) ? 0 : 8;
  const _Float16* p = s + (size_t)r * ld + k0 + off;
  union { v16h v; uint4 q[2]; } u;
  u.q[0] = *(const uint4*)(p);
  u.q[1] = *(const uint4*)(p + 16);
  return u.v;
}

// B-operand (32x16 f16): lane n (n<16) holds column n, K {k0..k0+7, k0+16..k0+23};
// lane n+16 holds column n, K {k0+8..k0+15, k0+24..k0+31}.  Source LDS stores B column n as
// row n (row-major, stride ld halves), so both chunks are contiguous.
__device__ __forceinline__ v16h load_b_tile(const _Float16* s, int col0, int k0,
                                            int ld, int lane) {
  int n   = col0 + (lane & 15);
  int off = (lane < 16) ? 0 : 8;
  const _Float16* p = s + (size_t)n * ld + k0 + off;
  union { v16h v; uint4 q[2]; } u;
  u.q[0] = *(const uint4*)(p);
  u.q[1] = *(const uint4*)(p + 16);
  return u.v;
}

// 16-lane-group reductions (xor masks 1..8 never cross the 16-lane halves,
// which hold distinct row groups in the WMMA C layout).
__device__ __forceinline__ float redmax16(float x) {
  x = fmaxf(x, __shfl_xor(x, 1, 32));
  x = fmaxf(x, __shfl_xor(x, 2, 32));
  x = fmaxf(x, __shfl_xor(x, 4, 32));
  x = fmaxf(x, __shfl_xor(x, 8, 32));
  return x;
}
__device__ __forceinline__ float redsum16(float x) {
  x += __shfl_xor(x, 1, 32);
  x += __shfl_xor(x, 2, 32);
  x += __shfl_xor(x, 4, 32);
  x += __shfl_xor(x, 8, 32);
  return x;
}

// Stage one 128x64 fp32 global tile into LDS as f16 (optionally also the
// 64x128 transpose), with optional scaling. 256 threads, float4 loads.
// Phase-split so all 8 global_load_b128 are in flight before any wait.
__device__ __forceinline__ void stage_tile(const float* __restrict__ g,
                                           _Float16* dst, _Float16* dstT,
                                           float scl, int tid) {
  float4 x[8];
  #pragma unroll
  for (int it = 0; it < 8; ++it) {
    int id4 = tid + it * 256;            // 2048 float4s = 128x64 floats
    int r   = id4 >> 4;
    int c   = (id4 & 15) << 2;
    x[it] = *(const float4*)(g + (size_t)r * D_MODEL + c);
  }
  #pragma unroll
  for (int it = 0; it < 8; ++it) {
    int id4 = tid + it * 256;
    int r   = id4 >> 4;
    int c   = (id4 & 15) << 2;
    _Float16 h0 = (_Float16)(x[it].x * scl);
    _Float16 h1 = (_Float16)(x[it].y * scl);
    _Float16 h2 = (_Float16)(x[it].z * scl);
    _Float16 h3 = (_Float16)(x[it].w * scl);
    union { _Float16 h[4]; uint2 u; } t;
    t.h[0] = h0; t.h[1] = h1; t.h[2] = h2; t.h[3] = h3;
    *(uint2*)(dst + (size_t)r * HD + c) = t.u;
    if (dstT) {
      dstT[(size_t)(c + 0) * BLK + r] = h0;
      dstT[(size_t)(c + 1) * BLK + r] = h1;
      dstT[(size_t)(c + 2) * BLK + r] = h2;
      dstT[(size_t)(c + 3) * BLK + r] = h3;
    }
  }
}

// Core per-(i,j) math for one wave's 16x128 row strip:
//   cS <- Pn = rowsoftmax(Q*scale @ K^T)   (f32)
//   cA <- dS = Pn * (dO @ V^T - rowdot)    (f32)
__device__ __forceinline__ void compute_pn_ds(const v16h* aQ, const v16h* aO,
                                              const _Float16* sK,
                                              const _Float16* sV,
                                              v8f* cS, v8f* cA, int lane) {
  #pragma unroll
  for (int t = 0; t < 8; ++t) {
    v8f c = {0, 0, 0, 0, 0, 0, 0, 0};
    c = wmma16(aQ[0], load_b_tile(sK, t * 16, 0, HD, lane), c);
    c = wmma16(aQ[1], load_b_tile(sK, t * 16, 32, HD, lane), c);
    cS[t] = c;
    v8f d = {0, 0, 0, 0, 0, 0, 0, 0};
    d = wmma16(aO[0], load_b_tile(sV, t * 16, 0, HD, lane), d);
    d = wmma16(aO[1], load_b_tile(sV, t * 16, 32, HD, lane), d);
    cA[t] = d;
  }
  #pragma unroll
  for (int v = 0; v < 8; ++v) {
    float mx = cS[0][v];
    #pragma unroll
    for (int t = 1; t < 8; ++t) mx = fmaxf(mx, cS[t][v]);
    mx = redmax16(mx);
    float l = 0.f;
    #pragma unroll
    for (int t = 0; t < 8; ++t) {
      float p = __expf(cS[t][v] - mx);
      cS[t][v] = p;
      l += p;
    }
    l = redsum16(l);
    float inv = 1.f / (l + 1e-12f);
    float rd  = 0.f;
    #pragma unroll
    for (int t = 0; t < 8; ++t) {
      float p = cS[t][v] * inv;
      cS[t][v] = p;
      rd += p * cA[t][v];
    }
    rd = redsum16(rd);
    #pragma unroll
    for (int t = 0; t < 8; ++t) cA[t][v] = cS[t][v] * (cA[t][v] - rd);
  }
}

// ---------------------------------------------------------------------------
// Kernel 1: one WG per (h, i) -> dQ_i  (loop over unmasked j)
// LDS: sQ sO sK sKT sV (16KB each) + sDS (32KB) = 112KB
// ---------------------------------------------------------------------------
__global__ void __launch_bounds__(256) dq_kernel(
    const float* __restrict__ q, const float* __restrict__ k,
    const float* __restrict__ v, const float* __restrict__ dO,
    const int* __restrict__ mask, float* __restrict__ dQ) {
  extern __shared__ _Float16 smem[];
  _Float16* sQ  = smem;
  _Float16* sO  = sQ + BLK * HD;
  _Float16* sK  = sO + BLK * HD;
  _Float16* sKT = sK + BLK * HD;
  _Float16* sV  = sKT + BLK * HD;
  _Float16* sDS = sV + BLK * HD;

  int tid = threadIdx.x, lane = tid & 31, wave = tid >> 5;
  int h = blockIdx.x >> 4, i = blockIdx.x & 15;

  stage_tile(q + (size_t)i * BLK * D_MODEL + h * HD, sQ, nullptr, SCALE, tid);
  stage_tile(dO + (size_t)i * BLK * D_MODEL + h * HD, sO, nullptr, 1.f, tid);
  __syncthreads();

  int p0 = wave * 16;
  v16h aQ[2], aO[2];
  aQ[0] = load_a_tile(sQ, p0, 0, HD, lane);
  aQ[1] = load_a_tile(sQ, p0, 32, HD, lane);
  aO[0] = load_a_tile(sO, p0, 0, HD, lane);
  aO[1] = load_a_tile(sO, p0, 32, HD, lane);

  v8f zero = {0, 0, 0, 0, 0, 0, 0, 0};
  v8f cQ[4];
  #pragma unroll
  for (int nt = 0; nt < 4; ++nt) cQ[nt] = zero;

  for (int j = 0; j < TBLK; ++j) {
    if (mask[i * TBLK + j] == 0) continue;
    if (j + 1 < TBLK) {
      __builtin_prefetch(k + (size_t)(j + 1) * BLK * D_MODEL + h * HD, 0, 0);
      __builtin_prefetch(v + (size_t)(j + 1) * BLK * D_MODEL + h * HD, 0, 0);
    }
    __syncthreads();
    stage_tile(k + (size_t)j * BLK * D_MODEL + h * HD, sK, sKT, 1.f, tid);
    stage_tile(v + (size_t)j * BLK * D_MODEL + h * HD, sV, nullptr, 1.f, tid);
    __syncthreads();

    v8f cS[8], cA[8];
    compute_pn_ds(aQ, aO, sK, sV, cS, cA, lane);

    // stage dS (row-major f16); wave-local rows p0..p0+15
    #pragma unroll
    for (int t = 0; t < 8; ++t) {
      int col = t * 16 + (lane & 15);
      int rb  = p0 + ((lane < 16) ? 0 : 8);
      #pragma unroll
      for (int vv = 0; vv < 8; ++vv)
        sDS[(size_t)(rb + vv) * BLK + col] = (_Float16)cA[t][vv];
    }
    // dQ_strip += dS(16x128) @ K(128x64)
    #pragma unroll
    for (int kt = 0; kt < 4; ++kt) {
      v16h a = load_a_tile(sDS, p0, kt * 32, BLK, lane);
      #pragma unroll
      for (int nt = 0; nt < 4; ++nt) {
        v16h b = load_b_tile(sKT, nt * 16, kt * 32, BLK, lane);
        cQ[nt] = wmma16(a, b, cQ[nt]);
      }
    }
  }

  #pragma unroll
  for (int nt = 0; nt < 4; ++nt) {
    int col = h * HD + nt * 16 + (lane & 15);
    int rb  = i * BLK + p0 + ((lane < 16) ? 0 : 8);
    #pragma unroll
    for (int vv = 0; vv < 8; ++vv)
      dQ[(size_t)(rb + vv) * D_MODEL + col] = cQ[nt][vv] * SCALE;
  }
}

// ---------------------------------------------------------------------------
// Kernel 2: one WG per (h, j) -> dK_j, dV_j  (loop over unmasked i)
// LDS: sK sV sQ sQT sO sOT (16KB each) + sPT sDST (32KB each) = 160KB
// ---------------------------------------------------------------------------
__global__ void __launch_bounds__(256) dkv_kernel(
    const float* __restrict__ q, const float* __restrict__ k,
    const float* __restrict__ v, const float* __restrict__ dO,
    const int* __restrict__ mask, float* __restrict__ dK,
    float* __restrict__ dV) {
  extern __shared__ _Float16 smem[];
  _Float16* sK   = smem;
  _Float16* sV   = sK + BLK * HD;
  _Float16* sQ   = sV + BLK * HD;
  _Float16* sQT  = sQ + BLK * HD;
  _Float16* sO   = sQT + BLK * HD;
  _Float16* sOT  = sO + BLK * HD;
  _Float16* sPT  = sOT + BLK * HD;   // Pn^T, 128x128
  _Float16* sDST = sPT + BLK * BLK;  // dS^T, 128x128

  int tid = threadIdx.x, lane = tid & 31, wave = tid >> 5;
  int h = blockIdx.x >> 4, j = blockIdx.x & 15;

  stage_tile(k + (size_t)j * BLK * D_MODEL + h * HD, sK, nullptr, 1.f, tid);
  stage_tile(v + (size_t)j * BLK * D_MODEL + h * HD, sV, nullptr, 1.f, tid);
  __syncthreads();

  int p0 = wave * 16;  // wave's strip: p-rows for S, q-rows for outputs
  v8f zero = {0, 0, 0, 0, 0, 0, 0, 0};
  v8f cK[4], cV[4];
  #pragma unroll
  for (int nt = 0; nt < 4; ++nt) { cK[nt] = zero; cV[nt] = zero; }

  for (int i = 0; i < TBLK; ++i) {
    if (mask[i * TBLK + j] == 0) continue;
    if (i + 1 < TBLK) {
      __builtin_prefetch(q + (size_t)(i + 1) * BLK * D_MODEL + h * HD, 0, 0);
      __builtin_prefetch(dO + (size_t)(i + 1) * BLK * D_MODEL + h * HD, 0, 0);
    }
    __syncthreads();
    stage_tile(q + (size_t)i * BLK * D_MODEL + h * HD, sQ, sQT, SCALE, tid);
    stage_tile(dO + (size_t)i * BLK * D_MODEL + h * HD, sO, sOT, 1.f, tid);
    __syncthreads();

    v16h aQ[2], aO[2];
    aQ[0] = load_a_tile(sQ, p0, 0, HD, lane);
    aQ[1] = load_a_tile(sQ, p0, 32, HD, lane);
    aO[0] = load_a_tile(sO, p0, 0, HD, lane);
    aO[1] = load_a_tile(sO, p0, 32, HD, lane);

    v8f cS[8], cA[8];
    compute_pn_ds(aQ, aO, sK, sV, cS, cA, lane);

    // stage Pn^T and dS^T: one b128 store each per tile per lane
    #pragma unroll
    for (int t = 0; t < 8; ++t) {
      int qc = t * 16 + (lane & 15);
      int pb = p0 + ((lane < 16) ? 0 : 8);
      union { _Float16 hh[8]; uint4 u; } tp, td;
      #pragma unroll
      for (int vv = 0; vv < 8; ++vv) {
        tp.hh[vv] = (_Float16)cS[t][vv];
        td.hh[vv] = (_Float16)cA[t][vv];
      }
      *(uint4*)(sPT + (size_t)qc * BLK + pb)  = tp.u;
      *(uint4*)(sDST + (size_t)qc * BLK + pb) = td.u;
    }
    __syncthreads();  // Pn^T/dS^T rows are written by all waves

    // dV_strip += Pn^T(16x128) @ dO(128x64); dK_strip += dS^T @ (Q*scale)
    #pragma unroll
    for (int kt = 0; kt < 4; ++kt) {
      v16h aP = load_a_tile(sPT, p0, kt * 32, BLK, lane);
      v16h aD = load_a_tile(sDST, p0, kt * 32, BLK, lane);
      #pragma unroll
      for (int nt = 0; nt < 4; ++nt) {
        v16h bO = load_b_tile(sOT, nt * 16, kt * 32, BLK, lane);
        cV[nt] = wmma16(aP, bO, cV[nt]);
        v16h bQ = load_b_tile(sQT, nt * 16, kt * 32, BLK, lane);
        cK[nt] = wmma16(aD, bQ, cK[nt]);
      }
    }
  }

  #pragma unroll
  for (int nt = 0; nt < 4; ++nt) {
    int col = h * HD + nt * 16 + (lane & 15);
    int rb  = j * BLK + p0 + ((lane < 16) ? 0 : 8);
    #pragma unroll
    for (int vv = 0; vv < 8; ++vv) {
      dK[(size_t)(rb + vv) * D_MODEL + col] = cK[nt][vv];  // scale folded into Q
      dV[(size_t)(rb + vv) * D_MODEL + col] = cV[nt][vv];
    }
  }
}

// ---------------------------------------------------------------------------
extern "C" void kernel_launch(void* const* d_in, const int* in_sizes, int n_in,
                              void* d_out, int out_size, void* d_ws,
                              size_t ws_size, hipStream_t stream) {
  (void)in_sizes; (void)n_in; (void)out_size; (void)d_ws; (void)ws_size;
  const float* q    = (const float*)d_in[0];
  const float* k    = (const float*)d_in[1];
  const float* v    = (const float*)d_in[2];
  const float* dO   = (const float*)d_in[3];
  const int*   mask = (const int*)d_in[4];

  float* out = (float*)d_out;
  float* dQ = out;
  float* dK = out + (size_t)2048 * 1024;
  float* dV = out + (size_t)2 * 2048 * 1024;

  const size_t lds1 = (size_t)(5 * BLK * HD + BLK * BLK) * 2;      // 112 KB
  const size_t lds2 = (size_t)(6 * BLK * HD + 2 * BLK * BLK) * 2;  // 160 KB

  dq_kernel<<<dim3(256), dim3(256), lds1, stream>>>(q, k, v, dO, mask, dQ);
  dkv_kernel<<<dim3(256), dim3(256), lds2, stream>>>(q, k, v, dO, mask, dK, dV);
}
